// MultiGAT_55035710931590
// MI455X (gfx1250) — compile-verified, matrix operand order checked by reading, not run on hardware
//
#include <hip/hip_runtime.h>
#include <hip/hip_bf16.h>

#define N_NODES 10000
#define N_EDGES 320000
#define CH      256      // H*C = 4*64, also IN_CH
#define HEADS   4
#define HC      64
#define NEG_SLOPE 0.2f

typedef __attribute__((ext_vector_type(16))) _Float16 v16h;
typedef __attribute__((ext_vector_type(8)))  _Float16 v8h;
typedef __attribute__((ext_vector_type(8)))  float    v8f;

union Frag16 { v16h v; v8h h[2]; };

// ---------------------------------------------------------------------------
// Converters
// ---------------------------------------------------------------------------
__global__ void f32_to_f16_kernel(const float* __restrict__ in,
                                  _Float16* __restrict__ out, int n) {
    int i = blockIdx.x * blockDim.x + threadIdx.x;
    if (i < n) out[i] = (_Float16)in[i];
}

// W [256,256] row-major (K x N) -> Wt [N][K] f16 (column-major of W)
__global__ void w_transpose_f16_kernel(const float* __restrict__ W,
                                       _Float16* __restrict__ Wt) {
    int i = blockIdx.x * blockDim.x + threadIdx.x;   // 65536
    int n = i & 255, k = i >> 8;
    Wt[n * CH + k] = (_Float16)W[k * CH + n];
}

__global__ void fill_kernel(float* __restrict__ p, float v, int n) {
    int i = blockIdx.x * blockDim.x + threadIdx.x;
    if (i < n) p[i] = v;
}

// ---------------------------------------------------------------------------
// WMMA GEMM: C[10000,256] = A[10000,256](f16) * B[256,256]
// B given transposed: Bt[n*256+k]. One wave -> one 16x16 tile, K in 8x32 steps.
// Fragment layout per CDNA5 ISA 7.12.2 (16-bit A 16x32): lane<16 holds row=lane,
// halves 0..7 -> K 0..7, halves 8..15 -> K 16..23; lanes 16..31 same rows, K+8.
// B fragment mirrors A with lane -> column (B effectively K x 16 col-major).
// ---------------------------------------------------------------------------
__global__ __launch_bounds__(256)
void gat_gemm_wmma_kernel(const _Float16* __restrict__ A,
                          const _Float16* __restrict__ Bt,
                          float* __restrict__ C) {
    const int lane = threadIdx.x & 31;
    const int wave = threadIdx.x >> 5;
    const int m0   = blockIdx.x * 16;                 // 625 * 16 = 10000 exact
    const int n0   = (blockIdx.y * 8 + wave) * 16;    // 2 * 8 * 16 = 256 exact
    const int half = lane >> 4;                       // 0 or 1
    const int ln   = lane & 15;
    const int kb   = half * 8;                        // K sub-offset per lane half

    const _Float16* arow = A  + (size_t)(m0 + ln) * CH;
    const _Float16* bcol = Bt + (size_t)(n0 + ln) * CH;

    v8f c = {};
#pragma unroll
    for (int kk = 0; kk < CH; kk += 32) {
        Frag16 a, b;
        a.h[0] = *(const v8h*)(arow + kk + kb);        // K = kk+kb .. +7
        a.h[1] = *(const v8h*)(arow + kk + kb + 16);   // K = kk+kb+16 .. +23
        b.h[0] = *(const v8h*)(bcol + kk + kb);
        b.h[1] = *(const v8h*)(bcol + kk + kb + 16);
        c = __builtin_amdgcn_wmma_f32_16x16x32_f16(
                /*neg_a=*/false, a.v, /*neg_b=*/false, b.v,
                /*c_mod=*/(short)0, c, /*reuse_a=*/false, /*reuse_b=*/false);
    }

    // D: VGPR r -> row r + 8*half, col = ln
    float* ccol = C + n0 + ln;
#pragma unroll
    for (int r = 0; r < 8; ++r) {
        int row = m0 + r + half * 8;
        ccol[(size_t)row * CH] = c[r];
    }
}

// ---------------------------------------------------------------------------
// es/ed: per (node, head) dot(h[n,head,:], a_src/a_dst[head,:])
// ---------------------------------------------------------------------------
__global__ void esed_kernel(const float* __restrict__ h,
                            const float* __restrict__ a_src,
                            const float* __restrict__ a_dst,
                            float* __restrict__ es, float* __restrict__ ed) {
    int i = blockIdx.x * blockDim.x + threadIdx.x;    // N_NODES*HEADS
    if (i >= N_NODES * HEADS) return;
    int hh = i & 3, node = i >> 2;
    const float* hp  = h + (size_t)node * CH + hh * HC;
    const float* asp = a_src + hh * HC;
    const float* adp = a_dst + hh * HC;
    float s = 0.f, d = 0.f;
#pragma unroll 8
    for (int c = 0; c < HC; ++c) { float hv = hp[c]; s += hv * asp[c]; d += hv * adp[c]; }
    es[i] = s; ed[i] = d;
}

// ---------------------------------------------------------------------------
// Edge softmax passes
// ---------------------------------------------------------------------------
__device__ inline float leaky(float v) { return v > 0.f ? v : NEG_SLOPE * v; }

__device__ inline void atomicMaxF32(float* addr, float v) {
    // classic monotonic float-bits trick; m initialized to -inf
    if (v >= 0.f) atomicMax((int*)addr, __float_as_int(v));
    else          atomicMin((unsigned int*)addr, __float_as_uint(v));
}

__global__ void edge_max_kernel(const int* __restrict__ src, const int* __restrict__ dst,
                                const float* __restrict__ es, const float* __restrict__ ed,
                                float* __restrict__ m) {
    int e = blockIdx.x * blockDim.x + threadIdx.x;
    if (e >= N_EDGES) return;
    int s = src[e], d = dst[e];
    const float4 sv = *(const float4*)(es + s * 4);
    const float4 dv = *(const float4*)(ed + d * 4);
    atomicMaxF32(m + d * 4 + 0, leaky(sv.x + dv.x));
    atomicMaxF32(m + d * 4 + 1, leaky(sv.y + dv.y));
    atomicMaxF32(m + d * 4 + 2, leaky(sv.z + dv.z));
    atomicMaxF32(m + d * 4 + 3, leaky(sv.w + dv.w));
}

__global__ void edge_sum_kernel(const int* __restrict__ src, const int* __restrict__ dst,
                                const float* __restrict__ es, const float* __restrict__ ed,
                                const float* __restrict__ m, float* __restrict__ ssum) {
    int e = blockIdx.x * blockDim.x + threadIdx.x;
    if (e >= N_EDGES) return;
    int s = src[e], d = dst[e];
    const float4 sv = *(const float4*)(es + s * 4);
    const float4 dv = *(const float4*)(ed + d * 4);
    const float4 mv = *(const float4*)(m + d * 4);
    atomicAdd(ssum + d * 4 + 0, __expf(leaky(sv.x + dv.x) - mv.x));
    atomicAdd(ssum + d * 4 + 1, __expf(leaky(sv.y + dv.y) - mv.y));
    atomicAdd(ssum + d * 4 + 2, __expf(leaky(sv.z + dv.z) - mv.z));
    atomicAdd(ssum + d * 4 + 3, __expf(leaky(sv.w + dv.w) - mv.w));
}

// One wave (32 lanes) per edge: lanes 0..3 hold per-head alpha, shfl-broadcast,
// then 8 coalesced gather + atomic-add pairs per lane (256 channels).
__global__ __launch_bounds__(256)
void edge_agg_kernel(const int* __restrict__ src, const int* __restrict__ dst,
                     const float* __restrict__ es, const float* __restrict__ ed,
                     const float* __restrict__ m, const float* __restrict__ ssum,
                     const float* __restrict__ h, float* __restrict__ agg) {
    int lane = threadIdx.x & 31;
    int wave = threadIdx.x >> 5;
    int e = blockIdx.x * 8 + wave;                    // 40000 blocks * 8 = 320000 exact
    if (e >= N_EDGES) return;
    int s = src[e], d = dst[e];

    int hh = lane & 3;
    float v = leaky(es[s * 4 + hh] + ed[d * 4 + hh]);
    float p = __expf(v - m[d * 4 + hh]);
    float al = p / (ssum[d * 4 + hh] + 1e-16f);
    float alpha[HEADS];
#pragma unroll
    for (int k = 0; k < HEADS; ++k) alpha[k] = __shfl(al, k, 32);

    const float* hs = h + (size_t)s * CH;
    float* od = agg + (size_t)d * CH;
#pragma unroll
    for (int j = 0; j < 8; ++j) {
        int idx = lane + j * 32;                      // contiguous across lanes
        atomicAdd(od + idx, alpha[idx >> 6] * hs[idx]);
    }
}

// ---------------------------------------------------------------------------
// Epilogue: layers 0..2: next_in = relu(agg + b) + x ; final: out = relu(agg+b+x)
// ---------------------------------------------------------------------------
__global__ void epilogue_kernel(const float* __restrict__ agg, const float* __restrict__ b,
                                const float* __restrict__ x0, float* __restrict__ out,
                                int final_layer) {
    int i = blockIdx.x * blockDim.x + threadIdx.x;
    if (i >= N_NODES * CH) return;
    float v = agg[i] + b[i & 255];
    if (final_layer) { v += x0[i]; out[i] = v > 0.f ? v : 0.f; }
    else             { v = v > 0.f ? v : 0.f; out[i] = v + x0[i]; }
}

// ---------------------------------------------------------------------------
// Host driver
// ---------------------------------------------------------------------------
extern "C" void kernel_launch(void* const* d_in, const int* in_sizes, int n_in,
                              void* d_out, int out_size, void* d_ws, size_t ws_size,
                              hipStream_t stream) {
    (void)in_sizes; (void)n_in; (void)out_size; (void)ws_size;

    const float* x   = (const float*)d_in[0];
    const int*   src = (const int*)d_in[1];
    const int*   dst = src + N_EDGES;

    const float* Ws[4] = {(const float*)d_in[2],  (const float*)d_in[6],
                          (const float*)d_in[10], (const float*)d_in[14]};
    const float* As[4] = {(const float*)d_in[3],  (const float*)d_in[7],
                          (const float*)d_in[11], (const float*)d_in[15]};
    const float* Ad[4] = {(const float*)d_in[4],  (const float*)d_in[8],
                          (const float*)d_in[12], (const float*)d_in[16]};
    const float* Bs[4] = {(const float*)d_in[5],  (const float*)d_in[9],
                          (const float*)d_in[13], (const float*)d_in[17]};

    // workspace carve-up (256B aligned)
    char* ws = (char*)d_ws;
    size_t off = 0;
    auto carve = [&](size_t bytes) -> void* {
        void* p = ws + off;
        off += (bytes + 255) & ~(size_t)255;
        return p;
    };
    _Float16* xh    = (_Float16*)carve((size_t)N_NODES * CH * 2);
    float*    hfeat = (float*)carve((size_t)N_NODES * CH * 4);
    float*    agg   = (float*)carve((size_t)N_NODES * CH * 4);
    float*    nin   = (float*)carve((size_t)N_NODES * CH * 4);
    float*    es    = (float*)carve((size_t)N_NODES * HEADS * 4);
    float*    ed    = (float*)carve((size_t)N_NODES * HEADS * 4);
    float*    mbuf  = (float*)carve((size_t)N_NODES * HEADS * 4);
    float*    ssum  = (float*)carve((size_t)N_NODES * HEADS * 4);
    _Float16* Wt    = (_Float16*)carve((size_t)4 * CH * CH * 2);

    const int NC  = N_NODES * CH;      // 2,560,000
    const int NH  = N_NODES * HEADS;   // 40,000
    const float NEG_INF = -__builtin_huge_valf();

    const float* cur = x;
    for (int l = 0; l < 4; ++l) {
        _Float16* Wtl = Wt + (size_t)l * CH * CH;

        w_transpose_f16_kernel<<<(CH * CH) / 256, 256, 0, stream>>>(Ws[l], Wtl);
        f32_to_f16_kernel<<<(NC + 255) / 256, 256, 0, stream>>>(cur, xh, NC);

        gat_gemm_wmma_kernel<<<dim3(N_NODES / 16, 2), 256, 0, stream>>>(xh, Wtl, hfeat);

        esed_kernel<<<(NH + 255) / 256, 256, 0, stream>>>(hfeat, As[l], Ad[l], es, ed);

        fill_kernel<<<(NC + 255) / 256, 256, 0, stream>>>(agg, 0.f, NC);
        fill_kernel<<<(NH + 255) / 256, 256, 0, stream>>>(mbuf, NEG_INF, NH);
        fill_kernel<<<(NH + 255) / 256, 256, 0, stream>>>(ssum, 0.f, NH);

        edge_max_kernel<<<(N_EDGES + 255) / 256, 256, 0, stream>>>(src, dst, es, ed, mbuf);
        edge_sum_kernel<<<(N_EDGES + 255) / 256, 256, 0, stream>>>(src, dst, es, ed, mbuf, ssum);
        edge_agg_kernel<<<N_EDGES / 8, 256, 0, stream>>>(src, dst, es, ed, mbuf, ssum, hfeat, agg);

        float* outp = (l == 3) ? (float*)d_out : nin;
        epilogue_kernel<<<(NC + 255) / 256, 256, 0, stream>>>(agg, Bs[l], x, outp, l == 3 ? 1 : 0);

        cur = nin;
    }
}
